// Sender_9887014716026
// MI455X (gfx1250) — matrix operand order repaired and checked by reading, not compiled, not run on hardware
//
#include <hip/hip_runtime.h>
#include <hip/hip_bf16.h>

// ---------------------------------------------------------------------------
// MI455X (gfx1250, wave32) implementation of the GRU "Sender".
// All matmuls run through v_wmma_f32_16x16x32_bf16 (fp32 accumulate).
// Logits (524 MB fp32) are the HBM-bound part; out_W in bf16 (32.8 MB) lives
// in the 192 MB L2 across all 16 decode steps.
// GEMM inner loop is software-pipelined (ping-pong fragment buffers) so
// global_load_b128 of iteration i+1 overlaps the 4 WMMAs of iteration i.
// ---------------------------------------------------------------------------

typedef __attribute__((ext_vector_type(16))) __bf16 v16bf;
typedef __attribute__((ext_vector_type(8)))  __bf16 v8bf;
typedef __attribute__((ext_vector_type(8)))  float  v8f;

static constexpr int BATCH = 256;
static constexpr int DIN   = 512;
static constexpr int DF    = 512;
static constexpr int H     = 512;
static constexpr int VOC   = 32000;
static constexpr int E     = 64;
static constexpr int LSEQ  = 16;

// ---------------------------- small utility kernels ------------------------

__global__ void k_f32_to_bf16(const float* __restrict__ src,
                              __bf16* __restrict__ dst, int n) {
  int i = blockIdx.x * blockDim.x + threadIdx.x;
  if (i < n) dst[i] = (__bf16)src[i];
}

__global__ void k_broadcast_sos(const float* __restrict__ sos,
                                __bf16* __restrict__ e, int n) {
  int i = blockIdx.x * blockDim.x + threadIdx.x;
  if (i < n) e[i] = (__bf16)sos[i & (E - 1)];
}

// ---------------------------- WMMA bf16 GEMM -------------------------------
// C[m,n] = sum_k A[m,k] * B[n,k] + bias[n]       (B given row-major [N,K])
// Block: 256 threads = 8 waves. WG tile 64(M) x 128(N); wave tile 16x64.
// grid.x = N/128, grid.y = M/64.  K must be a multiple of 64 (here 64 or 512).
__global__ void k_gemm_bf16_wmma(const __bf16* __restrict__ A, int lda,
                                 const __bf16* __restrict__ Bw, int ldb,
                                 const float* __restrict__ bias,
                                 float* __restrict__ C, long long ldc,
                                 long long cbase,
                                 __bf16* __restrict__ Cbf, int ldcbf,
                                 int K) {
  const int lane = threadIdx.x & 31;   // wave32 on gfx1250
  const int wave = threadIdx.x >> 5;
  const int m0 = blockIdx.y * 64 + (wave >> 1) * 16;
  const int n0 = blockIdx.x * 128 + (wave & 1) * 64;
  const int col  = lane & 15;          // M (for A) / N (for B,C) index in tile
  const int half = lane >> 4;          // K half selector

  v8f acc[4] = {};

  // A fragment (16x32 bf16): lane 0-15 -> M=lane, K=0..15;
  // lane 16-31 -> same M, K=16..31.
  const __bf16* aRow = A + (long long)(m0 + col) * lda + half * 16;
  // B fragment (32x16 bf16): lane 0-15 -> N=lane, K=0..15; lanes 16-31 K=16..31.
  const __bf16* bRow = Bw + (long long)(n0 + col) * ldb + half * 16;

  union AU { v16bf v; v8bf h[2]; };
  AU a0, a1;
  v16bf b0[4], b1[4];

  // Batched fragment loaders: issue all loads of a stage together so the
  // backend can clause them and overlap them with WMMAs of the other stage.
#define LOAD_A(dst, k0)                                                      \
  do {                                                                       \
    (dst).h[0] = *(const v8bf*)(aRow + (k0));                                \
    (dst).h[1] = *(const v8bf*)(aRow + (k0) + 8);                            \
  } while (0)
#define LOAD_B(dst, k0)                                                      \
  do {                                                                       \
    _Pragma("unroll")                                                        \
    for (int t = 0; t < 4; ++t)                                              \
      (dst)[t] = *(const v16bf*)(bRow + (long long)(t * 16) * ldb + (k0));   \
  } while (0)
#define WMMA4(aa, bb)                                                        \
  do {                                                                       \
    _Pragma("unroll")                                                        \
    for (int t = 0; t < 4; ++t)                                              \
      acc[t] = __builtin_amdgcn_wmma_f32_16x16x32_bf16(                      \
          false, (aa).v, false, (bb)[t], (short)0, acc[t], false, false);    \
  } while (0)

  // Prologue: stage 0.
  LOAD_A(a0, 0);
  LOAD_B(b0, 0);

  for (int k0 = 0; k0 < K; k0 += 64) {
    // Stage 1 loads overlap stage-0 WMMAs.
    LOAD_A(a1, k0 + 32);
    LOAD_B(b1, k0 + 32);
    // Prefetch a future K-chunk of B into the cache hierarchy
    // (gfx1250 global_prefetch_b8); B streams from L2 where bf16 out_W
    // (32.8 MB) is resident.
    if (k0 + 128 < K) __builtin_prefetch(bRow + k0 + 128, 0, 1);
    WMMA4(a0, b0);
    // Stage 0 loads of the next iteration overlap stage-1 WMMAs.
    if (k0 + 64 < K) {
      LOAD_A(a0, k0 + 64);
      LOAD_B(b0, k0 + 64);
    }
    WMMA4(a1, b1);
  }
#undef LOAD_A
#undef LOAD_B
#undef WMMA4

  // Epilogue: D layout (32-bit C/D 16x16): lane<16 -> M=r, lane>=16 -> M=8+r.
#pragma unroll
  for (int t = 0; t < 4; ++t) {
    const int n = n0 + t * 16 + col;
    const float bv = bias ? bias[n] : 0.0f;
#pragma unroll
    for (int r = 0; r < 8; ++r) {
      const int m = m0 + half * 8 + r;
      const float v = acc[t][r] + bv;
      C[cbase + (long long)m * ldc + n] = v;
      if (Cbf) Cbf[m * ldcbf + n] = (__bf16)v;
    }
  }
}

// ---------------------------- GRU gate math --------------------------------
__global__ void k_gru_gate(const float* __restrict__ gi,
                           const float* __restrict__ gh,
                           float* __restrict__ h,
                           __bf16* __restrict__ hbf) {
  int i = blockIdx.x * blockDim.x + threadIdx.x;
  if (i >= BATCH * H) return;
  int b = i >> 9;            // /H
  int j = i & (H - 1);
  const float* gib = gi + (long long)b * 3 * H;
  const float* ghb = gh + (long long)b * 3 * H;
  float r = 1.0f / (1.0f + __expf(-(gib[j] + ghb[j])));
  float z = 1.0f / (1.0f + __expf(-(gib[H + j] + ghb[H + j])));
  float n = tanhf(gib[2 * H + j] + r * ghb[2 * H + j]);
  float hn = (1.0f - z) * n + z * h[i];
  h[i]   = hn;
  hbf[i] = (__bf16)hn;
}

// -------------------- argmax over V + embedding gather ---------------------
// One block per batch row. First-occurrence tie-break like jnp.argmax.
__global__ void k_argmax_embed(const float* __restrict__ logits,
                               long long lstride, long long lbase,
                               const float* __restrict__ emb,
                               __bf16* __restrict__ e,
                               float* __restrict__ seq, int t) {
  __shared__ float sval[256];
  __shared__ int   sidx[256];
  const int b = blockIdx.x;
  const float* row = logits + lbase + (long long)b * lstride;
  float best = -3.402823466e38f;
  int   bi   = 0x7fffffff;
  for (int v = threadIdx.x; v < VOC; v += 256) {
    float x = row[v];
    if (x > best || (x == best && v < bi)) { best = x; bi = v; }
  }
  sval[threadIdx.x] = best;
  sidx[threadIdx.x] = bi;
  __syncthreads();
  for (int s = 128; s > 0; s >>= 1) {
    if (threadIdx.x < s) {
      float ov = sval[threadIdx.x + s]; int oi = sidx[threadIdx.x + s];
      float mv = sval[threadIdx.x];     int mi = sidx[threadIdx.x];
      if (ov > mv || (ov == mv && oi < mi)) {
        sval[threadIdx.x] = ov; sidx[threadIdx.x] = oi;
      }
    }
    __syncthreads();
  }
  const int sym = sidx[0];
  if (threadIdx.x == 0) seq[b * LSEQ + t] = (float)sym;
  if (threadIdx.x < E)
    e[b * E + threadIdx.x] = (__bf16)emb[(long long)sym * E + threadIdx.x];
}

// ---------------------------------------------------------------------------

extern "C" void kernel_launch(void* const* d_in, const int* in_sizes, int n_in,
                              void* d_out, int out_size, void* d_ws, size_t ws_size,
                              hipStream_t stream) {
  (void)in_sizes; (void)n_in; (void)out_size; (void)ws_size;
  const float* x     = (const float*)d_in[0];
  const float* enc_W = (const float*)d_in[1];
  const float* enc_b = (const float*)d_in[2];
  const float* in_W  = (const float*)d_in[3];
  const float* in_b  = (const float*)d_in[4];
  const float* W_ih  = (const float*)d_in[5];
  const float* W_hh  = (const float*)d_in[6];
  const float* b_ih  = (const float*)d_in[7];
  const float* b_hh  = (const float*)d_in[8];
  const float* out_W = (const float*)d_in[9];
  const float* out_b = (const float*)d_in[10];
  const float* emb   = (const float*)d_in[11];
  const float* sos   = (const float*)d_in[12];
  float* out = (float*)d_out;                 // [B*L] seq (as float) then [B,L,V] logits

  // ---- workspace carve-out (≈42 MB) ----
  size_t off = 0;
  auto wsp = [&](size_t bytes) -> void* {
    void* p = (char*)d_ws + off;
    off = (off + bytes + 255) & ~(size_t)255;
    return p;
  };
  __bf16* outW_bf = (__bf16*)wsp((size_t)VOC * H * 2);
  __bf16* Whh_bf  = (__bf16*)wsp((size_t)3 * H * H * 2);
  __bf16* Wih_bf  = (__bf16*)wsp((size_t)3 * H * E * 2);
  __bf16* encW_bf = (__bf16*)wsp((size_t)DF * DIN * 2);
  __bf16* inW_bf  = (__bf16*)wsp((size_t)H * DF * 2);
  __bf16* x_bf    = (__bf16*)wsp((size_t)BATCH * DIN * 2);
  __bf16* feat_bf = (__bf16*)wsp((size_t)BATCH * DF * 2);
  __bf16* h_bf    = (__bf16*)wsp((size_t)BATCH * H * 2);
  __bf16* e_bf    = (__bf16*)wsp((size_t)BATCH * E * 2);
  float*  feat_f  = (float*)wsp((size_t)BATCH * DF * 4);
  float*  h_f     = (float*)wsp((size_t)BATCH * H * 4);
  float*  gi      = (float*)wsp((size_t)BATCH * 3 * H * 4);
  float*  gh      = (float*)wsp((size_t)BATCH * 3 * H * 4);

  auto cvt = [&](const float* s, __bf16* d, int n) {
    k_f32_to_bf16<<<(n + 255) / 256, 256, 0, stream>>>(s, d, n);
  };

  // ---- one-time bf16 conversions (per launch; deterministic) ----
  cvt(out_W, outW_bf, VOC * H);
  cvt(W_hh,  Whh_bf,  3 * H * H);
  cvt(W_ih,  Wih_bf,  3 * H * E);
  cvt(enc_W, encW_bf, DF * DIN);
  cvt(in_W,  inW_bf,  H * DF);
  cvt(x,     x_bf,    BATCH * DIN);
  k_broadcast_sos<<<(BATCH * E + 255) / 256, 256, 0, stream>>>(sos, e_bf, BATCH * E);

  const dim3 blk(256);
  // feat = x @ enc_W^T + enc_b   (also emit bf16 copy for next GEMM)
  k_gemm_bf16_wmma<<<dim3(DF / 128, BATCH / 64), blk, 0, stream>>>(
      x_bf, DIN, encW_bf, DIN, enc_b, feat_f, DF, 0, feat_bf, DF, DIN);
  // h0 = feat @ in_W^T + in_b
  k_gemm_bf16_wmma<<<dim3(H / 128, BATCH / 64), blk, 0, stream>>>(
      feat_bf, DF, inW_bf, DF, in_b, h_f, H, 0, h_bf, H, DF);

  const long long seqN    = (long long)BATCH * LSEQ;       // 4096 floats
  const long long ldLogit = (long long)LSEQ * VOC;         // row stride in logits

  for (int t = 0; t < LSEQ; ++t) {
    // gi = e @ W_ih^T + b_ih      [256 x 1536], K=64
    k_gemm_bf16_wmma<<<dim3(3 * H / 128, BATCH / 64), blk, 0, stream>>>(
        e_bf, E, Wih_bf, E, b_ih, gi, 3 * H, 0, (__bf16*)nullptr, 0, E);
    // gh = h @ W_hh^T + b_hh      [256 x 1536], K=512
    k_gemm_bf16_wmma<<<dim3(3 * H / 128, BATCH / 64), blk, 0, stream>>>(
        h_bf, H, Whh_bf, H, b_hh, gh, 3 * H, 0, (__bf16*)nullptr, 0, H);
    // gate math -> new h (fp32 + bf16)
    k_gru_gate<<<(BATCH * H + 255) / 256, 256, 0, stream>>>(gi, gh, h_f, h_bf);
    // logits_t = h @ out_W^T + out_b, written straight into d_out at [b, t, :]
    k_gemm_bf16_wmma<<<dim3(VOC / 128, BATCH / 64), blk, 0, stream>>>(
        h_bf, H, outW_bf, H, out_b, out, ldLogit, seqN + (long long)t * VOC,
        (__bf16*)nullptr, 0, H);
    // greedy symbol + next embedding
    k_argmax_embed<<<BATCH, 256, 0, stream>>>(
        out, ldLogit, seqN + (long long)t * VOC, emb, e_bf, out, t);
  }
}
